// MoE_3393024164194
// MI455X (gfx1250) — matrix operand order repaired and checked by reading, not compiled, not run on hardware
//
#include <hip/hip_runtime.h>
#include <math.h>

#define H  1024
#define E  8
#define FF 4096
#define T  4096            // B*S tokens

typedef __attribute__((ext_vector_type(16))) _Float16 v16h;
typedef __attribute__((ext_vector_type(8)))  _Float16 v8h;
typedef __attribute__((ext_vector_type(8)))  float    v8f;

#define MT 32              // token tile (M)
#define KC 128             // FF chunk per phase
#define XP (H + 8)         // padded LDS stride (halves) for x tile
#define HP (KC + 8)        // padded LDS stride (halves) for hidden tile

#define WMMA_F16(a, b, c) \
  __builtin_amdgcn_wmma_f32_16x16x32_f16(false, (a), false, (b), (short)0, (c), false, false)

__device__ __forceinline__ float gelu_exact(float v) {
  return 0.5f * v * (1.0f + erff(v * 0.70710678118654752f));
}

// A fragment (16x32 f16) from LDS, rows [row0, row0+16). Contiguous k layout:
// lane m = lane&15 (+row0); halves 0..15 -> k = off16+{0..15},
// off16 = 16 for lanes 16..31. Two contiguous 16B runs -> ds_load_b128 pairs.
__device__ __forceinline__ v16h load_A_lds(const _Float16* base, int ldh, int row0,
                                           int k0, int lane) {
  const _Float16* p = base + (size_t)((lane & 15) + row0) * ldh + k0 + ((lane >> 4) & 1) * 16;
  const v8h lo = *(const v8h*)p;
  const v8h hi = *(const v8h*)(p + 8);
  v16h a;
#pragma unroll
  for (int h = 0; h < 8; ++h) { a[h] = lo[h]; a[8 + h] = hi[h]; }
  return a;
}

// B fragment (32x16) from pre-transposed f16 weights Wt[n][k] (k contiguous, row stride ldk).
// lane n = lane&15; lanes 16..31 take k+16..k+31 -> 2 contiguous 16B vector loads per lane;
// lane pairs (l, l+16) cover adjacent 32B of one row -> full 64B line utilization.
__device__ __forceinline__ v16h load_B_h16(const _Float16* __restrict__ Wt, int k0, int n0,
                                           int ldk, int lane) {
  const v8h* p = (const v8h*)(Wt + (size_t)(n0 + (lane & 15)) * ldk + k0 +
                              (((lane >> 4) & 1) << 4));
  const v8h lo = p[0], hi = p[1];
  v16h b;
#pragma unroll
  for (int h = 0; h < 8; ++h) { b[h] = lo[h]; b[8 + h] = hi[h]; }
  return b;
}

// Prefetch the cachelines a later load_B_h16(Wt, k0, n0, ...) will touch
// (32 lanes -> 16 rows x 64B; ~1KB per instruction, no VGPR dest, no LOADcnt).
__device__ __forceinline__ void prefetch_B(const _Float16* __restrict__ Wt, int k0, int n0,
                                           int ldk, int lane) {
  __builtin_prefetch(Wt + (size_t)(n0 + (lane & 15)) * ldk + k0 + (((lane >> 4) & 1) << 4),
                     0, 3);
}

// Fallback: B fragment straight from fp32 row-major W[k][n] (16 strided b32 loads + cvt).
__device__ __forceinline__ v16h load_B_f32(const float* __restrict__ W, int k0, int n0,
                                           int ldw, int lane) {
  v16h b;
  const float* p = W + (size_t)(k0 + ((lane >> 4) & 1) * 16) * ldw + n0 + (lane & 15);
#pragma unroll
  for (int h = 0; h < 16; ++h) b[h] = (_Float16)p[(size_t)h * ldw];
  return b;
}

// -------- weight pre-pass: fp32 [R][C] -> f16 transposed [C][R], tiled via LDS ---------
__global__ __launch_bounds__(256) void transpose_f32_to_f16(
    const float* __restrict__ src, _Float16* __restrict__ dst, int R, int C) {
  __shared__ float tile[32][33];
  const size_t mat = (size_t)R * C;
  const float* s = src + (size_t)blockIdx.z * mat;
  _Float16*    d = dst + (size_t)blockIdx.z * mat;
  const int c0 = blockIdx.x * 32, r0 = blockIdx.y * 32;
  const int tx = threadIdx.x & 31, ty = threadIdx.x >> 5;
#pragma unroll
  for (int i = 0; i < 4; ++i)
    tile[ty + i * 8][tx] = s[(size_t)(r0 + ty + i * 8) * C + c0 + tx];
  __syncthreads();
#pragma unroll
  for (int i = 0; i < 4; ++i)
    d[(size_t)(c0 + ty + i * 8) * R + r0 + tx] = (_Float16)tile[tx][ty + i * 8];
}

// ---------------- gating: sigmoid scores, top-2, normalized combine weights -------------
__global__ __launch_bounds__(256) void gate_topk_kernel(
    const float* __restrict__ x, const float* __restrict__ gw, const float* __restrict__ gb,
    int* __restrict__ cnt, int* __restrict__ tok_list, float* __restrict__ tok_wt) {
  const int wave = threadIdx.x >> 5;
  const int lane = threadIdx.x & 31;
  const int t = blockIdx.x * 8 + wave;
  if (t >= T) return;
  const float* xr = x + (size_t)t * H;
  float s[E];
#pragma unroll
  for (int e = 0; e < E; ++e) s[e] = 0.f;
  for (int h = lane; h < H; h += 32) {
    const float xv = xr[h];
#pragma unroll
    for (int e = 0; e < E; ++e) s[e] += xv * gw[e * H + h];
  }
#pragma unroll
  for (int e = 0; e < E; ++e) {
#pragma unroll
    for (int off = 16; off > 0; off >>= 1) s[e] += __shfl_xor(s[e], off, 32);
  }
  if (lane == 0) {
    float sc[E];
#pragma unroll
    for (int e = 0; e < E; ++e) sc[e] = 1.0f / (1.0f + expf(-(s[e] + gb[e])));
    int e0 = 0;
#pragma unroll
    for (int e = 1; e < E; ++e) if (sc[e] > sc[e0]) e0 = e;       // ties -> lower index
    int e1 = (e0 == 0) ? 1 : 0;
#pragma unroll
    for (int e = 0; e < E; ++e) if (e != e0 && sc[e] > sc[e1]) e1 = e;
    const float denom = sc[e0] + sc[e1] + 1e-6f;
    const float w0 = sc[e0] / denom, w1 = sc[e1] / denom;
    int p0 = atomicAdd(&cnt[e0], 1);
    tok_list[e0 * T + p0] = t; tok_wt[e0 * T + p0] = w0;
    int p1 = atomicAdd(&cnt[e1], 1);
    tok_list[e1 * T + p1] = t; tok_wt[e1 * T + p1] = w1;
  }
}

// ------------- fused expert FFN: y = (gelu(x@W1 + b1) @ W2 + b2) * weight ---------------
// grid.x = token tile, grid.y = expert id (E == shared expert). 8 waves / block.
// USE_H16: B operands from pre-transposed f16 weights in workspace.
template <bool USE_H16>
__global__ __launch_bounds__(256, 1) void moe_ffn_kernel(
    const float* __restrict__ x,
    const float* __restrict__ w1, const float* __restrict__ b1,
    const float* __restrict__ w2, const float* __restrict__ b2,
    const float* __restrict__ sw1, const float* __restrict__ sb1,
    const float* __restrict__ sw2, const float* __restrict__ sb2,
    const _Float16* __restrict__ w1t, const _Float16* __restrict__ w2t,
    const _Float16* __restrict__ sw1t, const _Float16* __restrict__ sw2t,
    const int* __restrict__ cnt, const int* __restrict__ tok_list,
    const float* __restrict__ tok_wt, float* __restrict__ out) {
  const int eid = blockIdx.y;
  const bool is_shared = (eid == E);
  const int m0 = blockIdx.x * MT;
  const float *W1 = nullptr, *W2 = nullptr, *B1, *B2;
  const _Float16 *W1T = nullptr, *W2T = nullptr;
  int nrows;
  if (is_shared) {
    nrows = T - m0; B1 = sb1; B2 = sb2;
    if (USE_H16) { W1T = sw1t; W2T = sw2t; } else { W1 = sw1; W2 = sw2; }
  } else {
    nrows = cnt[eid] - m0;
    B1 = b1 + (size_t)eid * FF;  B2 = b2 + (size_t)eid * H;
    if (USE_H16) { W1T = w1t + (size_t)eid * FF * H; W2T = w2t + (size_t)eid * H * FF; }
    else         { W1  = w1  + (size_t)eid * H * FF; W2  = w2  + (size_t)eid * FF * H; }
  }
  if (nrows <= 0) return;
  if (nrows > MT) nrows = MT;

  __shared__ _Float16 xs[MT * XP];   // x tile, f16
  __shared__ _Float16 hs[MT * HP];   // gelu(hidden) chunk, f16
  __shared__ int   s_tok[MT];
  __shared__ float s_wt[MT];

  const int tid  = threadIdx.x;
  const int wave = tid >> 5;
  const int lane = tid & 31;
  const int n_lo = lane & 15;
  const int hi8  = ((lane >> 4) & 1) * 8;

  if (tid < MT) {
    if (tid < nrows) {
      if (is_shared) { s_tok[tid] = m0 + tid; s_wt[tid] = 0.1f; }
      else { s_tok[tid] = tok_list[eid * T + m0 + tid];
             s_wt[tid]  = tok_wt [eid * T + m0 + tid]; }
    } else { s_tok[tid] = -1; s_wt[tid] = 0.f; }
  }
  __syncthreads();

  { // cooperative x-tile load: fp32 global -> f16 LDS, packed 16B LDS stores
    const int row = tid >> 3, seg = tid & 7;        // 8 segments * 128 cols
    const int tok = s_tok[row];
    _Float16* dst = xs + row * XP + seg * 128;
    if (tok >= 0) {
      const float4* src = (const float4*)(x + (size_t)tok * H + seg * 128);
#pragma unroll
      for (int i = 0; i < 16; ++i) {
        const float4 va = src[2 * i], vb = src[2 * i + 1];
        v8h h;
        h[0] = (_Float16)va.x; h[1] = (_Float16)va.y;
        h[2] = (_Float16)va.z; h[3] = (_Float16)va.w;
        h[4] = (_Float16)vb.x; h[5] = (_Float16)vb.y;
        h[6] = (_Float16)vb.z; h[7] = (_Float16)vb.w;
        *(v8h*)(dst + i * 8) = h;
      }
    } else {
      const v8h z = (v8h)(_Float16)0.f;
#pragma unroll
      for (int i = 0; i < 16; ++i) *(v8h*)(dst + i * 8) = z;
    }
  }
  __syncthreads();

  // y accumulators: wave owns output cols [wave*128, wave*128+128), both 16-row halves
  v8f yacc0[8], yacc1[8];
#pragma unroll
  for (int j = 0; j < 8; ++j) {
    const float bv = B2[wave * 128 + j * 16 + n_lo];
#pragma unroll
    for (int r = 0; r < 8; ++r) { yacc0[j][r] = bv; yacc1[j][r] = bv; }
  }

  for (int f0 = 0; f0 < FF; f0 += KC) {
    // Phase A: hidden[MT, KC] = gelu(x @ W1[:, f0:f0+KC] + b1); wave owns 16 cols.
    // B fragment software-pipelined; each k-step also prefetches one of the 32
    // W2 fragments this chunk's phase B will consume.
    {
      const int ncol = f0 + wave * 16;
      const float b1v = B1[ncol + n_lo];
      v8f h0, h1;
#pragma unroll
      for (int r = 0; r < 8; ++r) { h0[r] = b1v; h1[r] = b1v; }
      v16h bcur = USE_H16 ? load_B_h16(W1T, 0, ncol, H, lane)
                          : load_B_f32(W1, 0, ncol, FF, lane);
      for (int k0 = 0; k0 < H - 32; k0 += 32) {
        const v16h bnxt = USE_H16 ? load_B_h16(W1T, k0 + 32, ncol, H, lane)
                                  : load_B_f32(W1, k0 + 32, ncol, FF, lane);
        if (USE_H16) {
          const int pf = k0 >> 5;                   // 0..30 -> (j, kk) of phase B
          prefetch_B(W2T, f0 + ((pf >> 3) << 5), wave * 128 + (pf & 7) * 16, FF, lane);
        }
        const v16h a0 = load_A_lds(xs, XP, 0,  k0, lane);
        const v16h a1 = load_A_lds(xs, XP, 16, k0, lane);
        h0 = WMMA_F16(a0, bcur, h0);
        h1 = WMMA_F16(a1, bcur, h1);
        bcur = bnxt;
      }
      { // tail k-step (+ last phase-B prefetch)
        if (USE_H16) prefetch_B(W2T, f0 + 96, wave * 128 + 112, FF, lane);
        const v16h a0 = load_A_lds(xs, XP, 0,  H - 32, lane);
        const v16h a1 = load_A_lds(xs, XP, 16, H - 32, lane);
        h0 = WMMA_F16(a0, bcur, h0);
        h1 = WMMA_F16(a1, bcur, h1);
      }
      const int lc = wave * 16 + n_lo;
#pragma unroll
      for (int r = 0; r < 8; ++r) {
        hs[(r + hi8)      * HP + lc] = (_Float16)gelu_exact(h0[r]);
        hs[(r + hi8 + 16) * HP + lc] = (_Float16)gelu_exact(h1[r]);
      }
    }
    __syncthreads();
    // Phase B: y += hidden @ W2[f0:f0+KC, :]. Batch B fragments, then WMMA burst.
    // Each k-step also prefetches 1/4 of the next chunk's W1 slab (16 rows x 2KB).
    const bool pf_w1 = USE_H16 && (f0 + KC < FF);
    const int  ncol_nxt = f0 + KC + wave * 16;
#pragma unroll
    for (int k0 = 0; k0 < KC; k0 += 32) {
      v16h bfr[8];
#pragma unroll
      for (int j = 0; j < 8; ++j)
        bfr[j] = USE_H16 ? load_B_h16(W2T, f0 + k0, wave * 128 + j * 16, FF, lane)
                         : load_B_f32(W2, f0 + k0, wave * 128 + j * 16, H, lane);
      if (pf_w1) {
        const int base_k = (k0 >> 5) * 256;         // 4 steps x 8 x 32 halves = full H
#pragma unroll
        for (int i = 0; i < 8; ++i)
          prefetch_B(W1T, base_k + i * 32, ncol_nxt, H, lane);
      }
      const v16h a0 = load_A_lds(hs, HP, 0,  k0, lane);
      const v16h a1 = load_A_lds(hs, HP, 16, k0, lane);
#pragma unroll
      for (int j = 0; j < 8; ++j) {
        yacc0[j] = WMMA_F16(a0, bfr[j], yacc0[j]);
        yacc1[j] = WMMA_F16(a1, bfr[j], yacc1[j]);
      }
    }
    __syncthreads();
  }

  // scatter-add weighted results (C layout: lane n = lane&15, vgpr r -> row r + hi8)
#pragma unroll
  for (int j = 0; j < 8; ++j) {
    const int c = wave * 128 + j * 16 + n_lo;
#pragma unroll
    for (int r = 0; r < 8; ++r) {
      const int m = r + hi8;
      if (m < nrows) atomicAdd(&out[(size_t)s_tok[m] * H + c], s_wt[m] * yacc0[j][r]);
      const int m2 = m + 16;
      if (m2 < nrows) atomicAdd(&out[(size_t)s_tok[m2] * H + c], s_wt[m2] * yacc1[j][r]);
    }
  }
}

extern "C" void kernel_launch(void* const* d_in, const int* in_sizes, int n_in,
                              void* d_out, int out_size, void* d_ws, size_t ws_size,
                              hipStream_t stream) {
  const float* x   = (const float*)d_in[0];
  const float* gw  = (const float*)d_in[1];
  const float* gb  = (const float*)d_in[2];
  const float* w1  = (const float*)d_in[3];
  const float* b1  = (const float*)d_in[4];
  const float* w2  = (const float*)d_in[5];
  const float* b2  = (const float*)d_in[6];
  const float* sw1 = (const float*)d_in[7];
  const float* sb1 = (const float*)d_in[8];
  const float* sw2 = (const float*)d_in[9];
  const float* sb2 = (const float*)d_in[10];
  float* out = (float*)d_out;

  // workspace layout
  char* ws = (char*)d_ws;
  int*   cnt      = (int*)ws;                      // [16] (E used)
  int*   tok_list = cnt + 16;                      // [E*T]
  float* tok_wt   = (float*)(tok_list + E * T);    // [E*T]
  const size_t LIST_BYTES = 512 * 1024;            // lists, rounded up
  const size_t SZ_EXP = (size_t)H * FF * 2;        // one expert matrix in f16 bytes
  _Float16* w1t  = (_Float16*)(ws + LIST_BYTES);
  _Float16* w2t  = w1t  + (size_t)E * H * FF;
  _Float16* sw1t = w2t  + (size_t)E * H * FF;
  _Float16* sw2t = sw1t + (size_t)H * FF;
  const size_t needed = LIST_BYTES + SZ_EXP * (2 * E + 2);
  const bool use16 = (ws_size >= needed);

  hipMemsetAsync(cnt, 0, 16 * sizeof(int), stream);
  hipMemsetAsync(out, 0, (size_t)out_size * sizeof(float), stream);

  gate_topk_kernel<<<T / 8, 256, 0, stream>>>(x, gw, gb, cnt, tok_list, tok_wt);

  if (use16) {
    // fp32 [R][C] -> f16 [C][R]
    transpose_f32_to_f16<<<dim3(FF / 32, H / 32, E), 256, 0, stream>>>(w1,  w1t,  H,  FF);
    transpose_f32_to_f16<<<dim3(H / 32, FF / 32, E), 256, 0, stream>>>(w2,  w2t,  FF, H);
    transpose_f32_to_f16<<<dim3(FF / 32, H / 32, 1), 256, 0, stream>>>(sw1, sw1t, H,  FF);
    transpose_f32_to_f16<<<dim3(H / 32, FF / 32, 1), 256, 0, stream>>>(sw2, sw2t, FF, H);
    moe_ffn_kernel<true><<<dim3(T / MT, E + 1), 256, 0, stream>>>(
        x, w1, b1, w2, b2, sw1, sb1, sw2, sb2,
        w1t, w2t, sw1t, sw2t, cnt, tok_list, tok_wt, out);
  } else {
    moe_ffn_kernel<false><<<dim3(T / MT, E + 1), 256, 0, stream>>>(
        x, w1, b1, w2, b2, sw1, sb1, sw2, sb2,
        nullptr, nullptr, nullptr, nullptr, cnt, tok_list, tok_wt, out);
  }
}
